// MSARowAttentionWithPairBias_44375602102595
// MI455X (gfx1250) — matrix operand order repaired and checked by reading, not compiled, not run on hardware
//
#include <hip/hip_runtime.h>

// ---------------------------------------------------------------------------
// MSA Row Attention with Pair Bias — MI455X (gfx1250, wave32, WMMA f16)
//
// Pipeline (all f16 intermediates, f32 accumulate):
//  K0: transpose+convert weights   wq|wk|wv|wg -> Wt[N=1024][K=256] f16, wo -> Wto
//  K1: pair bias                   LN(z) . w_z -> bias[H][R][R] f32
//  K2: LayerNorm(m)                -> mln f16 [S*R][256]
//  K3: WMMA GEMM                   mln @ [wq|wk|wv|wg] -> q(scaled),k,v(transposed),g
//  K4: fused flash attention       per (s,h): K/V staged via async-to-LDS DMA,
//                                  QK^T + bias + mask, online softmax, PV,
//                                  gate multiply -> go f16
//  K5: WMMA GEMM                   go @ wo + bo -> out f32
// ---------------------------------------------------------------------------

typedef _Float16 f16;
typedef f16  v8h  __attribute__((ext_vector_type(8)));
typedef f16  v16h __attribute__((ext_vector_type(16)));
typedef float v8f __attribute__((ext_vector_type(8)));

#define S_   128
#define R_   256
#define CM_  256
#define CZ_  128
#define H_   8
#define HD_  32

__device__ __forceinline__ v8f wmma16x16x32(v16h a, v16h b, v8f c) {
  return __builtin_amdgcn_wmma_f32_16x16x32_f16(
      /*neg_a=*/false, a, /*neg_b=*/false, b,
      /*c_mod=*/(short)0, c, /*reuse_a=*/false, /*reuse_b=*/false);
}

// A fragment (16x32 f16, M x K): lane L (L%16 = M row), K half = (L/16)*8,
// VGPRs0-3 = K[koff..koff+7], VGPRs4-7 = K[koff+16..koff+23].
__device__ __forceinline__ v16h load_a_frag(const f16* __restrict__ base,
                                            int stride, int row0, int k0, int lane) {
  const int hl = lane >> 4, ln = lane & 15;
  const f16* p = base + (size_t)(row0 + ln) * stride + k0 + hl * 8;
  v8h lo = *(const v8h*)p;        // K = koff .. koff+7
  v8h hi = *(const v8h*)(p + 16); // K = koff+16 .. koff+23
  return __builtin_shufflevector(lo, hi, 0,1,2,3,4,5,6,7,8,9,10,11,12,13,14,15);
}

// B fragment (32x16 f16, K x N): lane L (L%16 = N col), K half = (L/16)*8,
// VGPRs0-3 = K[koff..koff+7], VGPRs4-7 = K[koff+16..koff+23].
__device__ __forceinline__ v16h load_b_frag(const f16* __restrict__ base,
                                            int stride, int col0, int k0, int lane) {
  const int hl = lane >> 4, ln = lane & 15;
  const f16* p = base + (size_t)(col0 + ln) * stride + k0 + hl * 8;
  v8h lo = *(const v8h*)p;
  v8h hi = *(const v8h*)(p + 16);
  return __builtin_shufflevector(lo, hi, 0,1,2,3,4,5,6,7,8,9,10,11,12,13,14,15);
}

// B fragment from a 32x32 f16 LDS tile (row = N col index, row stride 32).
__device__ __forceinline__ v16h load_b_lds(const f16* tile, int col0, int lane) {
  const int hl = lane >> 4, ln = lane & 15;
  const f16* p = tile + (col0 + ln) * 32 + hl * 8;
  v8h lo = *(const v8h*)p;
  v8h hi = *(const v8h*)(p + 16);
  return __builtin_shufflevector(lo, hi, 0,1,2,3,4,5,6,7,8,9,10,11,12,13,14,15);
}

// A fragment from LDS staging tile (16 rows x 32 cols f16, row stride 32).
__device__ __forceinline__ v16h load_a_lds(const f16* pst, int lane) {
  const int hl = lane >> 4, ln = lane & 15;
  const f16* p = pst + ln * 32 + hl * 8;
  v8h lo = *(const v8h*)p;
  v8h hi = *(const v8h*)(p + 16);
  return __builtin_shufflevector(lo, hi, 0,1,2,3,4,5,6,7,8,9,10,11,12,13,14,15);
}

// Async global->LDS copy, 16B per lane (GVS mode: SGPR base + per-lane offset).
// Tracked by ASYNCcnt; consumed after s_wait_asynccnt 0 + barrier.
__device__ __forceinline__ void async_copy_b128(unsigned lds_off,
                                                const void* sbase,
                                                int gbyte_off) {
  asm volatile("global_load_async_to_lds_b128 %0, %1, %2"
               :: "v"(lds_off), "v"(gbyte_off), "s"(sbase)
               : "memory");
}
__device__ __forceinline__ void wait_async0() {
  asm volatile("s_wait_asynccnt 0" ::: "memory");
}

__device__ __forceinline__ float wave_sum32(float v) {
#pragma unroll
  for (int o = 16; o >= 1; o >>= 1) v += __shfl_xor(v, o, 32);
  return v;
}
__device__ __forceinline__ float half_max16(float v) {
#pragma unroll
  for (int o = 8; o >= 1; o >>= 1) v = fmaxf(v, __shfl_xor(v, o, 16));
  return v;
}
__device__ __forceinline__ float half_sum16(float v) {
#pragma unroll
  for (int o = 8; o >= 1; o >>= 1) v += __shfl_xor(v, o, 16);
  return v;
}

// ---------------- K0: weight transpose + f32->f16 ---------------------------
__global__ __launch_bounds__(256) void k_cvt_w(
    const float* __restrict__ wq, const float* __restrict__ wk,
    const float* __restrict__ wv, const float* __restrict__ wg,
    const float* __restrict__ wo, f16* __restrict__ Wt, f16* __restrict__ Wto) {
  int idx = blockIdx.x * 256 + threadIdx.x;
  if (idx < 4 * 65536) {
    int reg = idx >> 16, e = idx & 65535;
    int n = e >> 8, k = e & 255;
    const float* w = (reg == 0) ? wq : (reg == 1) ? wk : (reg == 2) ? wv : wg;
    Wt[idx] = (f16)w[k * 256 + n];
  } else if (idx < 5 * 65536) {
    int e = idx & 65535;
    int n = e >> 8, k = e & 255;
    Wto[e] = (f16)wo[k * 256 + n];
  }
}

// ---------------- K1: pair bias = LN(z) . w_z -> [H][R][R] ------------------
__global__ __launch_bounds__(256) void k_pair_bias(
    const float* __restrict__ z, const float* __restrict__ gam,
    const float* __restrict__ bet, const float* __restrict__ wz,
    float* __restrict__ bias) {
  const int lane = threadIdx.x & 31, wave = threadIdx.x >> 5;
  const int idx = blockIdx.x * 8 + wave;  // flat (i,j), i*R + j
  const float* zr = z + (size_t)idx * CZ_;
  float4 u = *(const float4*)(zr + lane * 4);
  float x[4] = {u.x, u.y, u.z, u.w};
  float sm = wave_sum32(x[0] + x[1] + x[2] + x[3]);
  float mu = sm * (1.0f / CZ_);
  float sq = 0.0f;
#pragma unroll
  for (int c = 0; c < 4; ++c) { float d = x[c] - mu; sq += d * d; }
  sq = wave_sum32(sq);
  float rstd = rsqrtf(sq * (1.0f / CZ_) + 1e-5f);
  float zl[4];
#pragma unroll
  for (int c = 0; c < 4; ++c)
    zl[c] = (x[c] - mu) * rstd * gam[lane * 4 + c] + bet[lane * 4 + c];
#pragma unroll
  for (int hh = 0; hh < H_; ++hh) {
    float p = 0.0f;
#pragma unroll
    for (int c = 0; c < 4; ++c) p += zl[c] * wz[(lane * 4 + c) * H_ + hh];
    p = wave_sum32(p);
    if (lane == 0) bias[(size_t)hh * (R_ * R_) + idx] = p;
  }
}

// ---------------- K2: LayerNorm(m) -> f16 -----------------------------------
__global__ __launch_bounds__(256) void k_ln_m(
    const float* __restrict__ m, const float* __restrict__ gam,
    const float* __restrict__ bet, f16* __restrict__ mln) {
  const int lane = threadIdx.x & 31, wave = threadIdx.x >> 5;
  const int row = blockIdx.x * 8 + wave;
  const float* mr = m + (size_t)row * CM_;
  float4 u0 = *(const float4*)(mr + lane * 8);
  float4 u1 = *(const float4*)(mr + lane * 8 + 4);
  float x[8] = {u0.x, u0.y, u0.z, u0.w, u1.x, u1.y, u1.z, u1.w};
  float sm = 0.0f;
#pragma unroll
  for (int e = 0; e < 8; ++e) sm += x[e];
  sm = wave_sum32(sm);
  float mu = sm * (1.0f / CM_);
  float sq = 0.0f;
#pragma unroll
  for (int e = 0; e < 8; ++e) { float d = x[e] - mu; sq += d * d; }
  sq = wave_sum32(sq);
  float rstd = rsqrtf(sq * (1.0f / CM_) + 1e-5f);
  v8h o;
#pragma unroll
  for (int e = 0; e < 8; ++e)
    o[e] = (f16)((x[e] - mu) * rstd * gam[lane * 8 + e] + bet[lane * 8 + e]);
  *(v8h*)(mln + (size_t)row * CM_ + lane * 8) = o;
}

// ---------------- K3: WMMA GEMM  mln @ [wq|wk|wv|wg] ------------------------
// q pre-scaled by 1/sqrt(HD); v stored transposed [S,H,HD,R]; g = sigmoid(.+bg)
__global__ __launch_bounds__(256) void k_gemm_qkvg(
    const f16* __restrict__ mln, const f16* __restrict__ Wt,
    const float* __restrict__ bg, f16* __restrict__ qb, f16* __restrict__ kb,
    f16* __restrict__ vt, f16* __restrict__ gb) {
  const int lane = threadIdx.x & 31, wave = threadIdx.x >> 5;
  const int m0 = blockIdx.x * 128 + (wave >> 1) * 32;
  const int n0 = blockIdx.y * 64 + (wave & 1) * 32;
  v8f c[2][2] = {};
  for (int k0 = 0; k0 < CM_; k0 += 32) {
    v16h a0 = load_a_frag(mln, CM_, m0, k0, lane);
    v16h a1 = load_a_frag(mln, CM_, m0 + 16, k0, lane);
    v16h b0 = load_b_frag(Wt, CM_, n0, k0, lane);
    v16h b1 = load_b_frag(Wt, CM_, n0 + 16, k0, lane);
    c[0][0] = wmma16x16x32(a0, b0, c[0][0]);
    c[0][1] = wmma16x16x32(a0, b1, c[0][1]);
    c[1][0] = wmma16x16x32(a1, b0, c[1][0]);
    c[1][1] = wmma16x16x32(a1, b1, c[1][1]);
  }
  const int hl = lane >> 4, ln = lane & 15;
  const float qscale = 0.17677669529663689f;  // 1/sqrt(32)
#pragma unroll
  for (int tm = 0; tm < 2; ++tm)
#pragma unroll
    for (int tn = 0; tn < 2; ++tn)
#pragma unroll
      for (int r = 0; r < 8; ++r) {
        int row = m0 + tm * 16 + r + 8 * hl;     // global (s*R + i)
        int n = n0 + tn * 16 + ln;               // 0..1023
        float val = c[tm][tn][r];
        int reg = n >> 8, cc = n & 255;
        if (reg == 0) {
          qb[(size_t)row * 256 + cc] = (f16)(val * qscale);
        } else if (reg == 1) {
          kb[(size_t)row * 256 + cc] = (f16)val;
        } else if (reg == 2) {
          int s = row >> 8, i = row & 255;
          int hh = cc >> 5, d = cc & 31;
          vt[(size_t)(((s << 3) + hh) * 32 + d) * 256 + i] = (f16)val;
        } else {
          float gv = 1.0f / (1.0f + __expf(-(val + bg[cc])));
          gb[(size_t)row * 256 + cc] = (f16)gv;
        }
      }
}

// ---------------- K4: fused flash attention per (s,h) -----------------------
__global__ __launch_bounds__(256) void k_attn(
    const f16* __restrict__ qb, const f16* __restrict__ kbuf,
    const f16* __restrict__ vt, const float* __restrict__ bias,
    const float* __restrict__ mask, const f16* __restrict__ gb,
    f16* __restrict__ gob) {
  __shared__ float maskb[R_];
  __shared__ __align__(16) f16 pst_all[8 * 512];  // 16x32 f16 per wave
  __shared__ __align__(16) f16 ktile[32 * 32];    // [key][dim] f16, async-staged
  __shared__ __align__(16) f16 vtile[32 * 32];    // [dim][key] f16, async-staged
  const int s = blockIdx.x >> 3, h = blockIdx.x & 7;
  const int tid = threadIdx.x;
  const int lane = tid & 31, wave = tid >> 5;
  const int hl = lane >> 4, ln = lane & 15;

  maskb[tid] = 1e9f * (mask[s * R_ + tid] - 1.0f);

  f16* pst = pst_all + wave * 512;
  const int i0 = wave * 32;                 // 32 query rows per wave
  const int qrow = s * R_ + i0;

  v16h qa[2];
  qa[0] = load_a_frag(qb, CM_, qrow, h * HD_, lane);
  qa[1] = load_a_frag(qb, CM_, qrow + 16, h * HD_, lane);

  v8f O[2][2] = {};
  float mrow[2][8], lrow[2][8];
#pragma unroll
  for (int t = 0; t < 2; ++t)
#pragma unroll
    for (int r = 0; r < 8; ++r) { mrow[t][r] = -3.0e38f; lrow[t][r] = 0.0f; }

  const float* biash = bias + (size_t)h * (R_ * R_);

  // Async staging setup: 256 threads x 16B = 4KB per key block.
  // tid < 128: K rows  (key = tid>>2, 4x16B chunks of 64B row)
  // tid >=128: V^T rows (dim = (tid-128)>>2)
  const int chunk = tid & 3;
  const int krow = tid >> 2;           // 0..31 for tid<128
  const int vrow = (tid - 128) >> 2;   // 0..31 for tid>=128
  // uniform SGPR bases (byte pointers)
  const char* kbase = (const char*)kbuf + ((size_t)(s * R_) * CM_ + h * HD_) * 2;
  const char* vbase = (const char*)vt + ((size_t)((s << 3) + h) * HD_) * CM_ * 2;
  const unsigned k_lds = (unsigned)(size_t)(void*)ktile;
  const unsigned v_lds = (unsigned)(size_t)(void*)vtile;

  __syncthreads();

  for (int j0 = 0; j0 < R_; j0 += 32) {
    // ---- stage K/V block into LDS via async DMA (one copy per workgroup) ----
    if (tid < 128) {
      async_copy_b128(k_lds + krow * 64 + chunk * 16, kbase,
                      (j0 + krow) * (CM_ * 2) + chunk * 16);
    } else {
      async_copy_b128(v_lds + vrow * 64 + chunk * 16, vbase,
                      vrow * (CM_ * 2) + j0 * 2 + chunk * 16);
    }
    wait_async0();
    __syncthreads();

    v16h kb0 = load_b_lds(ktile, 0, lane);
    v16h kb1 = load_b_lds(ktile, 16, lane);
    v16h vb0 = load_b_lds(vtile, 0, lane);
    v16h vb1 = load_b_lds(vtile, 16, lane);

#pragma unroll
    for (int t = 0; t < 2; ++t) {
      v8f zero = {};
      v8f S0 = wmma16x16x32(qa[t], kb0, zero);
      v8f S1 = wmma16x16x32(qa[t], kb1, zero);
#pragma unroll
      for (int r = 0; r < 8; ++r) {
        int i = i0 + t * 16 + r + 8 * hl;
        S0[r] += biash[(size_t)i * R_ + j0 + ln] + maskb[j0 + ln];
        S1[r] += biash[(size_t)i * R_ + j0 + 16 + ln] + maskb[j0 + 16 + ln];
      }
#pragma unroll
      for (int r = 0; r < 8; ++r) {
        float mx = half_max16(fmaxf(S0[r], S1[r]));
        float nm = fmaxf(mrow[t][r], mx);
        float alpha = __expf(mrow[t][r] - nm);
        mrow[t][r] = nm;
        S0[r] = __expf(S0[r] - nm);
        S1[r] = __expf(S1[r] - nm);
        float ps = half_sum16(S0[r] + S1[r]);
        lrow[t][r] = lrow[t][r] * alpha + ps;
        O[t][0][r] *= alpha;
        O[t][1][r] *= alpha;
      }
      // P (C layout) -> LDS row-major -> A fragment layout
#pragma unroll
      for (int r = 0; r < 8; ++r) {
        int row = r + 8 * hl;
        pst[row * 32 + ln] = (f16)S0[r];
        pst[row * 32 + 16 + ln] = (f16)S1[r];
      }
      asm volatile("s_wait_dscnt 0" ::: "memory");  // wave-wide stage visible
      v16h pa = load_a_lds(pst, lane);
      O[t][0] = wmma16x16x32(pa, vb0, O[t][0]);
      O[t][1] = wmma16x16x32(pa, vb1, O[t][1]);
    }
    __syncthreads();  // all waves done reading K/V tile before next overwrite
  }

  // epilogue: 1/l normalization + gate multiply, store f16
#pragma unroll
  for (int t = 0; t < 2; ++t)
#pragma unroll
    for (int r = 0; r < 8; ++r) {
      float inv = 1.0f / lrow[t][r];
      size_t i = (size_t)(s * R_ + i0 + t * 16 + r + 8 * hl);
      int c0 = h * HD_ + ln;
      float g0 = (float)gb[i * 256 + c0];
      float g1 = (float)gb[i * 256 + c0 + 16];
      gob[i * 256 + c0] = (f16)(O[t][0][r] * inv * g0);
      gob[i * 256 + c0 + 16] = (f16)(O[t][1][r] * inv * g1);
    }
}

// ---------------- K5: WMMA GEMM  go @ wo + bo -> f32 out --------------------
__global__ __launch_bounds__(256) void k_gemm_out(
    const f16* __restrict__ go, const f16* __restrict__ Wto,
    const float* __restrict__ bo, float* __restrict__ out) {
  const int lane = threadIdx.x & 31, wave = threadIdx.x >> 5;
  const int m0 = blockIdx.x * 128 + (wave >> 1) * 32;
  const int n0 = blockIdx.y * 64 + (wave & 1) * 32;
  v8f c[2][2] = {};
  for (int k0 = 0; k0 < CM_; k0 += 32) {
    v16h a0 = load_a_frag(go, CM_, m0, k0, lane);
    v16h a1 = load_a_frag(go, CM_, m0 + 16, k0, lane);
    v16h b0 = load_b_frag(Wto, CM_, n0, k0, lane);
    v16h b1 = load_b_frag(Wto, CM_, n0 + 16, k0, lane);
    c[0][0] = wmma16x16x32(a0, b0, c[0][0]);
    c[0][1] = wmma16x16x32(a0, b1, c[0][1]);
    c[1][0] = wmma16x16x32(a1, b0, c[1][0]);
    c[1][1] = wmma16x16x32(a1, b1, c[1][1]);
  }
  const int hl = lane >> 4, ln = lane & 15;
#pragma unroll
  for (int tm = 0; tm < 2; ++tm)
#pragma unroll
    for (int tn = 0; tn < 2; ++tn)
#pragma unroll
      for (int r = 0; r < 8; ++r) {
        int row = m0 + tm * 16 + r + 8 * hl;
        int n = n0 + tn * 16 + ln;
        out[(size_t)row * CM_ + n] = c[tm][tn][r] + bo[n];
      }
}

// ---------------------------------------------------------------------------
extern "C" void kernel_launch(void* const* d_in, const int* in_sizes, int n_in,
                              void* d_out, int out_size, void* d_ws,
                              size_t ws_size, hipStream_t stream) {
  const float* m      = (const float*)d_in[0];
  const float* z      = (const float*)d_in[1];
  const float* mask   = (const float*)d_in[2];
  const float* ln_m_g = (const float*)d_in[3];
  const float* ln_m_b = (const float*)d_in[4];
  const float* ln_z_g = (const float*)d_in[5];
  const float* ln_z_b = (const float*)d_in[6];
  const float* w_z    = (const float*)d_in[7];
  const float* wq     = (const float*)d_in[8];
  const float* wk     = (const float*)d_in[9];
  const float* wv     = (const float*)d_in[10];
  const float* wg     = (const float*)d_in[11];
  const float* bg     = (const float*)d_in[12];
  const float* wo     = (const float*)d_in[13];
  const float* bo     = (const float*)d_in[14];

  const size_t NROW = (size_t)S_ * R_;          // 32768
  const size_t MELEM = NROW * CM_;              // 8388608

  char* ws = (char*)d_ws;
  f16*   Wt   = (f16*)ws;   ws += (size_t)4 * 65536 * sizeof(f16);
  f16*   Wto  = (f16*)ws;   ws += (size_t)65536 * sizeof(f16);
  float* bias = (float*)ws; ws += (size_t)H_ * R_ * R_ * sizeof(float);
  f16*   mln  = (f16*)ws;   ws += MELEM * sizeof(f16);
  f16*   qb   = (f16*)ws;   ws += MELEM * sizeof(f16);
  f16*   kb   = (f16*)ws;   ws += MELEM * sizeof(f16);
  f16*   vt   = (f16*)ws;   ws += MELEM * sizeof(f16);
  f16*   gb   = (f16*)ws;   ws += MELEM * sizeof(f16);
  f16*   gob  = (f16*)ws;   ws += MELEM * sizeof(f16);

  k_cvt_w<<<1280, 256, 0, stream>>>(wq, wk, wv, wg, wo, Wt, Wto);
  k_pair_bias<<<(R_ * R_) / 8, 256, 0, stream>>>(z, ln_z_g, ln_z_b, w_z, bias);
  k_ln_m<<<NROW / 8, 256, 0, stream>>>(m, ln_m_g, ln_m_b, mln);
  {
    dim3 g((unsigned)(NROW / 128), 1024 / 64);
    k_gemm_qkvg<<<g, 256, 0, stream>>>(mln, Wt, bg, qb, kb, vt, gb);
  }
  k_attn<<<S_ * H_, 256, 0, stream>>>(qb, kb, vt, bias, mask, gb, gob);
  {
    dim3 g((unsigned)(NROW / 128), CM_ / 64);
    k_gemm_out<<<g, 256, 0, stream>>>(gob, Wto, bo, (float*)d_out);
  }
}
